// GAT_module_5342939316930
// MI455X (gfx1250) — matrix operand order repaired and checked by reading, not compiled, not run on hardware
//
#include <hip/hip_runtime.h>
#include <hip/hip_bf16.h>

#define NN 100000
#define EE 1600000
#define HEADS 4
#define CDIM 32
#define FOUT 128      // HEADS*CDIM
#define INDIM 128
#define NEG_SLOPE 0.2f

typedef __attribute__((ext_vector_type(16))) __bf16 v16bf;
typedef __attribute__((ext_vector_type(8)))  float  v8f;

// ---------------------------------------------------------------------------
// Kernel 1: pack [W_src | W_dst] (each 128x128 f32, row-major KxF) into bf16
// swizzled exactly as the CDNA5 WMMA B-operand expects:
// flat index = ((t*4 + s)*32 + lane)*16 + e
//   t = 16-col tile (0..15), s = 32-K step (0..3), lane = 0..31, e = 0..15
//   K = s*32 + (lane>>4)*8 + (e&7) + (e>>3)*16 ;  col = t*16 + (lane&15)
// ---------------------------------------------------------------------------
__global__ void pack_w_kernel(const float* __restrict__ Wsrc,
                              const float* __restrict__ Wdst,
                              __bf16* __restrict__ Wp) {
  int i = blockIdx.x * blockDim.x + threadIdx.x;
  if (i >= 16 * 4 * 32 * 16) return;
  int e = i & 15;
  int l = (i >> 4) & 31;
  int s = (i >> 9) & 3;
  int t = i >> 11;
  int k = s * 32 + (l >> 4) * 8 + (e & 7) + (e >> 3) * 16;
  int col = t * 16 + (l & 15);
  float w = (col < FOUT) ? Wsrc[k * FOUT + col] : Wdst[k * FOUT + (col - FOUT)];
  Wp[i] = (__bf16)w;
}

// ---------------------------------------------------------------------------
// Kernel 2: init accumulators + copy edge_index into output slot
// ---------------------------------------------------------------------------
__global__ void init_kernel(float* __restrict__ out_x,
                            float* __restrict__ out_edge,
                            const int* __restrict__ ei,
                            float* __restrict__ amax,
                            float* __restrict__ denom) {
  size_t i = (size_t)blockIdx.x * blockDim.x + threadIdx.x;
  if (i < (size_t)NN * FOUT) out_x[i] = 0.0f;
  if (i < (size_t)NN * HEADS) {
    amax[i]  = -__builtin_inff();
    denom[i] = 0.0f;
  }
  if (i < (size_t)2 * EE) out_edge[i] = (float)ei[i];
}

// ---------------------------------------------------------------------------
// Kernel 3: H = x @ [W_src | W_dst] via v_wmma_f32_16x16x32_bf16.
// Block = 256 threads = 8 waves; block handles a 16-row tile of x.
// Wave w computes col tiles {w, w+8} of the 256 concatenated output cols.
// A layout (16-bit A 16x32): lane l holds row M=l&15; K runs
//   [s*32 + (l>>4)*8 .. +7] and [s*32 + 16 + (l>>4)*8 .. +7].
// ---------------------------------------------------------------------------
__global__ __launch_bounds__(256) void gemm_kernel(
    const float*  __restrict__ x,
    const __bf16* __restrict__ Wp,
    float* __restrict__ h_src,
    float* __restrict__ h_dst) {
  const int lane = threadIdx.x & 31;
  const int wave = threadIdx.x >> 5;
  const int m    = lane & 15;
  const int hi   = lane >> 4;
  const int row0 = blockIdx.x * 16;

  const float* xr = x + (size_t)(row0 + m) * INDIM;

  v16bf a[4];
#pragma unroll
  for (int s = 0; s < 4; ++s) {
    const int kb = s * 32 + hi * 8;
    float4 q0 = *(const float4*)(xr + kb);
    float4 q1 = *(const float4*)(xr + kb + 4);
    float4 q2 = *(const float4*)(xr + kb + 16);
    float4 q3 = *(const float4*)(xr + kb + 20);
    a[s][0]  = (__bf16)q0.x; a[s][1]  = (__bf16)q0.y;
    a[s][2]  = (__bf16)q0.z; a[s][3]  = (__bf16)q0.w;
    a[s][4]  = (__bf16)q1.x; a[s][5]  = (__bf16)q1.y;
    a[s][6]  = (__bf16)q1.z; a[s][7]  = (__bf16)q1.w;
    a[s][8]  = (__bf16)q2.x; a[s][9]  = (__bf16)q2.y;
    a[s][10] = (__bf16)q2.z; a[s][11] = (__bf16)q2.w;
    a[s][12] = (__bf16)q3.x; a[s][13] = (__bf16)q3.y;
    a[s][14] = (__bf16)q3.z; a[s][15] = (__bf16)q3.w;
  }

#pragma unroll
  for (int tt = 0; tt < 2; ++tt) {
    const int t = wave + tt * 8;        // col tile 0..15
    v8f acc = {};
#pragma unroll
    for (int s = 0; s < 4; ++s) {
      v16bf b = *(const v16bf*)(Wp + (((size_t)(t * 4 + s)) * 32 + lane) * 16);
      acc = __builtin_amdgcn_wmma_f32_16x16x32_bf16(
          /*neg_a=*/false, a[s], /*neg_b=*/false, b,
          /*c_mod=*/(short)0, acc, /*reuse_a=*/false, /*reuse_b=*/false);
    }
    // D layout: VGPR r -> M = r + 8*hi ; N = lane&15
    const int n = lane & 15;
    float* outp = tt ? h_dst : h_src;            // tt==0 -> cols 0..127 (src)
    const int col = t * 16 + n - tt * FOUT;
#pragma unroll
    for (int r = 0; r < 8; ++r) {
      const int gr = row0 + hi * 8 + r;
      outp[(size_t)gr * FOUT + col] = acc[r];
    }
  }
}

// ---------------------------------------------------------------------------
// Kernel 4: per-node attention logits  a_{src,dst}[n,h] = <h[n,h,:], att[h,:]>
// ---------------------------------------------------------------------------
__global__ void logits_kernel(const float* __restrict__ h_src,
                              const float* __restrict__ h_dst,
                              const float* __restrict__ att_src,
                              const float* __restrict__ att_dst,
                              float* __restrict__ a_src,
                              float* __restrict__ a_dst) {
  int i = blockIdx.x * blockDim.x + threadIdx.x;
  if (i >= NN * HEADS) return;
  int n = i >> 2, h = i & 3;
  const float* hs = h_src + (size_t)n * FOUT + h * CDIM;
  const float* hd = h_dst + (size_t)n * FOUT + h * CDIM;
  const float* as = att_src + h * CDIM;
  const float* ad = att_dst + h * CDIM;
  float s1 = 0.f, s2 = 0.f;
#pragma unroll
  for (int c = 0; c < CDIM; c += 4) {
    float4 u = *(const float4*)(hs + c);
    float4 v = *(const float4*)(hd + c);
    float4 p = *(const float4*)(as + c);
    float4 q = *(const float4*)(ad + c);
    s1 += u.x * p.x + u.y * p.y + u.z * p.z + u.w * p.w;
    s2 += v.x * q.x + v.y * q.y + v.z * q.z + v.w * q.w;
  }
  a_src[i] = s1;
  a_dst[i] = s2;
}

// ---------------------------------------------------------------------------
// float atomic max via int/uint bit trick (works with -inf init)
// ---------------------------------------------------------------------------
__device__ __forceinline__ void atomicMaxF(float* addr, float value) {
  if (value >= 0.f) atomicMax((int*)addr, __float_as_int(value));
  else              atomicMin((unsigned int*)addr, __float_as_uint(value));
}

// ---------------------------------------------------------------------------
// Kernel 5: per-edge logits + leaky-relu + segment max over dst
// ---------------------------------------------------------------------------
__global__ void edge_logit_kernel(const int* __restrict__ src,
                                  const int* __restrict__ dst,
                                  const float* __restrict__ a_src,
                                  const float* __restrict__ a_dst,
                                  float* __restrict__ attn,
                                  float* __restrict__ amax) {
  int i = blockIdx.x * blockDim.x + threadIdx.x;
  if (i >= EE * HEADS) return;
  int e = i >> 2, h = i & 3;
  int s = src[e], d = dst[e];
  float al = a_src[s * HEADS + h] + a_dst[d * HEADS + h];
  al = (al >= 0.f) ? al : NEG_SLOPE * al;
  attn[i] = al;
  atomicMaxF(&amax[d * HEADS + h], al);
}

// ---------------------------------------------------------------------------
// Kernel 6: exp(alpha - max) + segment sum
// ---------------------------------------------------------------------------
__global__ void edge_exp_kernel(const int* __restrict__ dst,
                                const float* __restrict__ amax,
                                float* __restrict__ attn,
                                float* __restrict__ denom) {
  int i = blockIdx.x * blockDim.x + threadIdx.x;
  if (i >= EE * HEADS) return;
  int e = i >> 2, h = i & 3;
  int d = dst[e];
  float ex = __expf(attn[i] - amax[d * HEADS + h]);
  attn[i] = ex;
  atomicAdd(&denom[d * HEADS + h], ex);
}

// ---------------------------------------------------------------------------
// Kernel 7: weighted scatter-add of messages. One wave per edge; lane c
// handles channel c of all 4 heads (fully coalesced gather + atomic add).
// Also finalizes attn = ex / (denom + eps) into the output slot.
// ---------------------------------------------------------------------------
__global__ __launch_bounds__(256) void scatter_kernel(
    const int* __restrict__ src,
    const int* __restrict__ dst,
    const float* __restrict__ h_src,
    const float* __restrict__ denom,
    float* __restrict__ attn,
    float* __restrict__ out) {
  int e = blockIdx.x * 8 + (threadIdx.x >> 5);
  int lane = threadIdx.x & 31;
  if (e >= EE) return;
  int s = src[e], d = dst[e];
  const float* hs = h_src + (size_t)s * FOUT;
  float* orow = out + (size_t)d * FOUT;
  float w[HEADS];
#pragma unroll
  for (int h = 0; h < HEADS; ++h) {
    float ex = attn[(size_t)e * HEADS + h];
    float dn = denom[d * HEADS + h];
    w[h] = ex / (dn + 1e-16f);
  }
  if (lane < HEADS) attn[(size_t)e * HEADS + lane] = w[lane];
#pragma unroll
  for (int h = 0; h < HEADS; ++h) {
    atomicAdd(&orow[h * CDIM + lane], w[h] * hs[h * CDIM + lane]);
  }
}

// ---------------------------------------------------------------------------
// Kernel 8: bias + rowwise LayerNorm + LeakyReLU. One wave per node.
// ---------------------------------------------------------------------------
__global__ __launch_bounds__(256) void ln_kernel(float* __restrict__ out,
                                                 const float* __restrict__ bias,
                                                 const float* __restrict__ gamma,
                                                 const float* __restrict__ beta) {
  int node = blockIdx.x * 8 + (threadIdx.x >> 5);
  int lane = threadIdx.x & 31;
  if (node >= NN) return;
  float* row = out + (size_t)node * FOUT;
  float4 v  = *(float4*)(row + lane * 4);
  float4 bb = *(const float4*)(bias + lane * 4);
  v.x += bb.x; v.y += bb.y; v.z += bb.z; v.w += bb.w;

  float sum = v.x + v.y + v.z + v.w;
#pragma unroll
  for (int off = 16; off; off >>= 1) sum += __shfl_xor(sum, off, 32);
  float mu = sum * (1.0f / FOUT);

  float dx = v.x - mu, dy = v.y - mu, dz = v.z - mu, dw = v.w - mu;
  float sq = dx * dx + dy * dy + dz * dz + dw * dw;
#pragma unroll
  for (int off = 16; off; off >>= 1) sq += __shfl_xor(sq, off, 32);
  float rstd = rsqrtf(sq * (1.0f / FOUT) + 1e-5f);

  float4 g  = *(const float4*)(gamma + lane * 4);
  float4 be = *(const float4*)(beta + lane * 4);
  float4 o;
  o.x = dx * rstd * g.x + be.x;
  o.y = dy * rstd * g.y + be.y;
  o.z = dz * rstd * g.z + be.z;
  o.w = dw * rstd * g.w + be.w;
  o.x = (o.x >= 0.f) ? o.x : NEG_SLOPE * o.x;
  o.y = (o.y >= 0.f) ? o.y : NEG_SLOPE * o.y;
  o.z = (o.z >= 0.f) ? o.z : NEG_SLOPE * o.z;
  o.w = (o.w >= 0.f) ? o.w : NEG_SLOPE * o.w;
  *(float4*)(row + lane * 4) = o;
}

// ---------------------------------------------------------------------------
extern "C" void kernel_launch(void* const* d_in, const int* in_sizes, int n_in,
                              void* d_out, int out_size, void* d_ws, size_t ws_size,
                              hipStream_t stream) {
  const float* x        = (const float*)d_in[0];
  // d_in[1] edge_attr: unused (with_edge='N')
  const int*   ei       = (const int*)d_in[2];   // [2, E]
  // d_in[3] batch: unused (per-graph LN == rowwise LN)
  const float* Wsrc     = (const float*)d_in[4];
  const float* Wdst     = (const float*)d_in[5];
  const float* att_src  = (const float*)d_in[6];
  const float* att_dst  = (const float*)d_in[7];
  const float* bias     = (const float*)d_in[8];
  const float* ln_gamma = (const float*)d_in[9];
  const float* ln_beta  = (const float*)d_in[10];

  const int* src = ei;
  const int* dst = ei + EE;

  // Output layout: [x_after: N*128][edge_index: 2E][attn: E*H]
  float* out_x    = (float*)d_out;
  float* out_edge = out_x + (size_t)NN * FOUT;
  float* out_attn = out_edge + (size_t)2 * EE;

  // Workspace layout
  char*   ws    = (char*)d_ws;
  __bf16* Wp    = (__bf16*)ws;                              // 64 KB swizzled B
  float*  h_src = (float*)(ws + 65536);
  float*  h_dst = h_src + (size_t)NN * FOUT;
  float*  a_src = h_dst + (size_t)NN * FOUT;
  float*  a_dst = a_src + (size_t)NN * HEADS;
  float*  amax  = a_dst + (size_t)NN * HEADS;
  float*  denom = amax  + (size_t)NN * HEADS;

  pack_w_kernel<<<(16 * 4 * 32 * 16) / 256, 256, 0, stream>>>(Wsrc, Wdst, Wp);

  size_t initN = (size_t)NN * FOUT;  // largest region touched by init
  init_kernel<<<(unsigned)((initN + 255) / 256), 256, 0, stream>>>(
      out_x, out_edge, ei, amax, denom);

  gemm_kernel<<<NN / 16, 256, 0, stream>>>(x, Wp, h_src, h_dst);

  logits_kernel<<<(NN * HEADS + 255) / 256, 256, 0, stream>>>(
      h_src, h_dst, att_src, att_dst, a_src, a_dst);

  edge_logit_kernel<<<(EE * HEADS + 255) / 256, 256, 0, stream>>>(
      src, dst, a_src, a_dst, out_attn, amax);

  edge_exp_kernel<<<(EE * HEADS + 255) / 256, 256, 0, stream>>>(
      dst, amax, out_attn, denom);

  scatter_kernel<<<EE / 8, 256, 0, stream>>>(
      src, dst, h_src, denom, out_attn, out_x);

  ln_kernel<<<(NN + 7) / 8, 256, 0, stream>>>(out_x, bias, ln_gamma, ln_beta);
}
